// Windowed3DAttentionLayer_52974126629360
// MI455X (gfx1250) — matrix operand order repaired and checked by reading, not compiled, not run on hardware
//
#include <hip/hip_runtime.h>

// ---------------------------------------------------------------------------
// CDNA5 / gfx1250 Swin-style 3D windowed attention block.
// All GEMMs run on v_wmma_f32_16x16x32_f16 (f16 in, f32 accumulate).
// GEMM tiles: BM=128, BN=128, BK=32, double-buffered LDS fed by
// GLOBAL_LOAD_ASYNC_TO_LDS_B128 (ASYNCcnt-tracked), synchronized with
// s_wait_asynccnt + workgroup barriers.
// ---------------------------------------------------------------------------

typedef __attribute__((ext_vector_type(16))) _Float16 v16h;
typedef __attribute__((ext_vector_type(8)))  float    v8f;
typedef __attribute__((ext_vector_type(4)))  _Float16 h4;
typedef int v4i __attribute__((vector_size(4 * sizeof(int))));

#define TOKENS 65536   // 2 * 32^3
#define CDIM   512
#define HIDDIM 2048

#if __has_builtin(__builtin_amdgcn_global_load_async_to_lds_b128) && \
    __has_builtin(__builtin_amdgcn_s_wait_asynccnt)
#define USE_ASYNC 1
#else
#define USE_ASYNC 0
#endif

// 16-byte global -> LDS copy: async DMA if available, else direct.
__device__ __forceinline__ void cp16(const _Float16* g, _Float16* l) {
#if USE_ASYNC
  __builtin_amdgcn_global_load_async_to_lds_b128(
      (__attribute__((address_space(1))) v4i*)g,
      (__attribute__((address_space(3))) v4i*)l, 0, 0);
#else
  *(uint4*)l = *(const uint4*)g;
#endif
}
__device__ __forceinline__ void wait_async_0() {
#if USE_ASYNC
  __builtin_amdgcn_s_wait_asynccnt(0);
#endif
}
__device__ __forceinline__ void wait_async_4() {
#if USE_ASYNC
  __builtin_amdgcn_s_wait_asynccnt(4);
#endif
}

// token index in shifted-window order -> token index in natural (b,d,h,w)
// order (roll(-2,-2,-2) + 4x4x4 window partition; same map serves as the
// window-reverse + unshift since it is a bijection we invert by writing).
__device__ __forceinline__ int map_src(int m) {
  int b    = m >> 15;
  int r    = m & 32767;
  int winb = r >> 6;
  int n    = r & 63;
  int wd = winb >> 6;
  int wh = (winb >> 3) & 7;
  int ww = winb & 7;
  int d = ((wd << 2) + (n >> 4)       + 2) & 31;
  int h = ((wh << 2) + ((n >> 2) & 3) + 2) & 31;
  int w = ((ww << 2) + (n & 3)        + 2) & 31;
  return ((b * 32 + d) * 32 + h) * 32 + w;
}

// --- WMMA fragment loaders (CDNA5 ISA 7.12.2 16-bit layouts) -----------------
__device__ __forceinline__ v16h frag_a(const _Float16* tile, int ldk) {
  int lane = threadIdx.x & 31;
  const _Float16* p = tile + (lane & 15) * ldk + ((lane >> 4) << 3);
  v16h r;
  ((uint4*)&r)[0] = *(const uint4*)(p);
  ((uint4*)&r)[1] = *(const uint4*)(p + 16);
  return r;
}
__device__ __forceinline__ v16h frag_b(const _Float16* tileT, int ldk) {
  int lane = threadIdx.x & 31;
  const _Float16* p = tileT + (lane & 15) * ldk + ((lane >> 4) << 3);
  v16h r;
  ((uint4*)&r)[0] = *(const uint4*)(p);
  ((uint4*)&r)[1] = *(const uint4*)(p + 16);
  return r;
}
__device__ __forceinline__ v8f wmma_f16(v16h a, v16h b, v8f c) {
  return __builtin_amdgcn_wmma_f32_16x16x32_f16(false, a, false, b,
                                                (short)0, c, false, false);
}

// --- weight convert + transpose: WT[n*K+k] = (f16) W[k*N+n] ------------------
__global__ __launch_bounds__(256) void wconv_kernel(const float* __restrict__ W,
                                                    _Float16* __restrict__ WT,
                                                    int K, int N) {
  int idx = blockIdx.x * 256 + threadIdx.x;
  if (idx >= K * N) return;
  int n = idx / K;
  int k = idx - n * K;
  WT[idx] = (_Float16)W[(size_t)k * N + n];
}

// --- LayerNorm (one wave per token), optional shifted-window gather ----------
__global__ __launch_bounds__(256) void ln_kernel(const float* __restrict__ in,
                                                 const float* __restrict__ gam,
                                                 const float* __restrict__ bet,
                                                 _Float16* __restrict__ outh,
                                                 int mapped) {
  int wv = threadIdx.x >> 5, lane = threadIdx.x & 31;
  int m = blockIdx.x * 8 + wv;
  int src = mapped ? map_src(m) : m;
  const float4* xr = (const float4*)(in + (size_t)src * CDIM);
  float4 vals[4];
  float s = 0.f, ss = 0.f;
#pragma unroll
  for (int i = 0; i < 4; i++) {
    float4 v = xr[i * 32 + lane];
    vals[i] = v;
    s  += v.x + v.y + v.z + v.w;
    ss += v.x * v.x + v.y * v.y + v.z * v.z + v.w * v.w;
  }
#pragma unroll
  for (int off = 16; off > 0; off >>= 1) {
    s  += __shfl_xor(s, off, 32);
    ss += __shfl_xor(ss, off, 32);
  }
  float mean = s * (1.f / CDIM);
  float var  = ss * (1.f / CDIM) - mean * mean;
  float rstd = rsqrtf(var + 1e-5f);
  h4* orow = (h4*)(outh + (size_t)m * CDIM);
#pragma unroll
  for (int i = 0; i < 4; i++) {
    float4 v  = vals[i];
    float4 g4 = ((const float4*)gam)[i * 32 + lane];
    float4 b4 = ((const float4*)bet)[i * 32 + lane];
    h4 o;
    o.x = (_Float16)((v.x - mean) * rstd * g4.x + b4.x);
    o.y = (_Float16)((v.y - mean) * rstd * g4.y + b4.y);
    o.z = (_Float16)((v.z - mean) * rstd * g4.z + b4.z);
    o.w = (_Float16)((v.w - mean) * rstd * g4.w + b4.w);
    orow[i * 32 + lane] = o;
  }
}

// --- Tiled WMMA GEMM: out = A[M,K] @ WT^T + bias, templated epilogue ---------
// BM=128, BN=128, BK=32; 256 threads = 8 waves in 2x4, wave tile 64x32
// (4x2 WMMA, 8 v_wmma per K-step for 6 fragment loads). LDS double-buffered,
// filled by async global->LDS copies overlapping the current K-step's math.
// EPI 0: f16 store (QKV). 1: GELU->f16 (FFN1).
// EPI 2: proj: fp32, residual = x at mapped index, write to mapped row (xo).
// EPI 3: ffn2: fp32, residual = xo at same row, write d_out.
template <int EPI>
__global__ __launch_bounds__(256) void gemm_ep(const _Float16* __restrict__ A,
                                               const _Float16* __restrict__ WT,
                                               const float* __restrict__ bias,
                                               void* __restrict__ outp,
                                               int K, int Ntot,
                                               const float* __restrict__ res) {
  __shared__ alignas(16) _Float16 As[2][128 * 32];
  __shared__ alignas(16) _Float16 Bs[2][128 * 32];
  int t = threadIdx.x;
  int wv = t >> 5;
  int wm = wv >> 2, wn = wv & 3;       // 2 x 4 wave grid
  int mBase = blockIdx.y * 128;
  int nBase = blockIdx.x * 128;
  v8f acc[4][2] = {};

  // per-thread tile-slice source rows/cols (16B granularity, 2 rounds each)
  int e0 = t * 8, e1 = t * 8 + 2048;
  int ar0 = e0 >> 5, ac0 = e0 & 31;
  int ar1 = e1 >> 5, ac1 = e1 & 31;

  int niter = K >> 5;
  // prologue: stage K-slice 0 into buffer 0
  cp16(A  + (size_t)(mBase + ar0) * K + ac0, &As[0][e0]);
  cp16(A  + (size_t)(mBase + ar1) * K + ac1, &As[0][e1]);
  cp16(WT + (size_t)(nBase + ar0) * K + ac0, &Bs[0][e0]);
  cp16(WT + (size_t)(nBase + ar1) * K + ac1, &Bs[0][e1]);

  for (int it = 0; it < niter; ++it) {
    int cur = it & 1;
    if (it + 1 < niter) {              // stage next slice while we compute
      int nb = cur ^ 1, k1 = (it + 1) << 5;
      cp16(A  + (size_t)(mBase + ar0) * K + k1 + ac0, &As[nb][e0]);
      cp16(A  + (size_t)(mBase + ar1) * K + k1 + ac1, &As[nb][e1]);
      cp16(WT + (size_t)(nBase + ar0) * K + k1 + ac0, &Bs[nb][e0]);
      cp16(WT + (size_t)(nBase + ar1) * K + k1 + ac1, &Bs[nb][e1]);
      wait_async_4();                  // oldest 4 (current buffer) done
    } else {
      wait_async_0();
    }
    __syncthreads();
    v16h af[4], bf[2];
#pragma unroll
    for (int tm = 0; tm < 4; tm++)
      af[tm] = frag_a(&As[cur][(wm * 64 + tm * 16) * 32], 32);
#pragma unroll
    for (int tn = 0; tn < 2; tn++)
      bf[tn] = frag_b(&Bs[cur][(wn * 32 + tn * 16) * 32], 32);
#pragma unroll
    for (int tm = 0; tm < 4; tm++)
#pragma unroll
      for (int tn = 0; tn < 2; tn++)
        acc[tm][tn] = wmma_f16(af[tm], bf[tn], acc[tm][tn]);
    __syncthreads();                   // readers done before next overwrite
  }

  int lane = t & 31;
  int nIn = lane & 15;
  int mIn = (lane >> 4) * 8;           // C/D: vgpr j -> M = mIn + j, N = nIn
#pragma unroll
  for (int tm = 0; tm < 4; tm++)
#pragma unroll
    for (int tn = 0; tn < 2; tn++) {
      v8f a = acc[tm][tn];
      int gn = nBase + wn * 32 + tn * 16 + nIn;
      float bcol = bias[gn];
#pragma unroll
      for (int j = 0; j < 8; j++) {
        int gm = mBase + wm * 64 + tm * 16 + mIn + j;
        float val = a[j] + bcol;
        if constexpr (EPI == 0) {
          ((_Float16*)outp)[(size_t)gm * Ntot + gn] = (_Float16)val;
        } else if constexpr (EPI == 1) {
          float ge = 0.5f * val * (1.f + erff(val * 0.70710678118654752f));
          ((_Float16*)outp)[(size_t)gm * Ntot + gn] = (_Float16)ge;
        } else if constexpr (EPI == 2) {
          size_t idx = (size_t)map_src(gm) * CDIM + gn;  // reverse + unshift
          ((float*)outp)[idx] = val + res[idx];          // xo = x + proj_out
        } else {
          size_t idx = (size_t)gm * CDIM + gn;
          ((float*)outp)[idx] = val + res[idx];          // out = xo + ffn
        }
      }
    }
}

// --- attention: one block per (window, head); N=64 tokens, hd=64 -------------
__global__ __launch_bounds__(256) void attn_kernel(const _Float16* __restrict__ qkv,
                                                   _Float16* __restrict__ attout) {
  __shared__ alignas(16) _Float16 qs[64 * 64];    // q, reused for P (f16)
  __shared__ alignas(16) _Float16 ks[64 * 64];    // k as [token][dim] == B^T
  __shared__ alignas(16) _Float16 vts[64 * 64];   // v transposed: [dim][token]
  __shared__ alignas(16) float    Ss[64 * 64];    // logits fp32
  int win = blockIdx.x >> 3, head = blockIdx.x & 7;
  int t = threadIdx.x;
  size_t base = (size_t)win * 64 * 1536 + head * 64;
#pragma unroll
  for (int cch = 0; cch < 2; cch++) {
    int e = t * 8 + cch * 2048;
    int row = e >> 6, col = e & 63;
    size_t g = base + (size_t)row * 1536 + col;
    cp16(qkv + g,       qs + e);                  // async q
    cp16(qkv + g + 512, ks + e);                  // async k
    _Float16 tmp[8];                              // v: transpose on the way in
    *(uint4*)tmp = *(const uint4*)(qkv + g + 1024);
#pragma unroll
    for (int ii = 0; ii < 8; ii++) vts[(col + ii) * 64 + row] = tmp[ii];
  }
  wait_async_0();
  __syncthreads();
  int wv = t >> 5, lane = t & 31;
  int sm = (wv >> 1) * 16, sn = (wv & 1) * 32;    // wave: 16x32 of 64x64
  int nIn = lane & 15, mIn = (lane >> 4) * 8;
  {
    v8f s0 = {}, s1 = {};
#pragma unroll
    for (int k0 = 0; k0 < 64; k0 += 32) {
      v16h a  = frag_a(qs + sm * 64 + k0, 64);
      v16h b0 = frag_b(ks + sn * 64 + k0, 64);
      v16h b1 = frag_b(ks + (sn + 16) * 64 + k0, 64);
      s0 = wmma_f16(a, b0, s0);
      s1 = wmma_f16(a, b1, s1);
    }
#pragma unroll
    for (int j = 0; j < 8; j++) {
      Ss[(sm + mIn + j) * 64 + sn + nIn]      = s0[j] * 0.125f;
      Ss[(sm + mIn + j) * 64 + sn + 16 + nIn] = s1[j] * 0.125f;
    }
  }
  __syncthreads();
  for (int rr = 0; rr < 8; rr++) {                // softmax: 8 rows per wave
    int row = wv * 8 + rr;
    float x0 = Ss[row * 64 + lane], x1 = Ss[row * 64 + lane + 32];
    float mx = fmaxf(x0, x1);
#pragma unroll
    for (int off = 16; off > 0; off >>= 1) mx = fmaxf(mx, __shfl_xor(mx, off, 32));
    float e0 = expf(x0 - mx), e1 = expf(x1 - mx);
    float sum = e0 + e1;
#pragma unroll
    for (int off = 16; off > 0; off >>= 1) sum += __shfl_xor(sum, off, 32);
    float inv = 1.f / sum;
    qs[row * 64 + lane]      = (_Float16)(e0 * inv);
    qs[row * 64 + lane + 32] = (_Float16)(e1 * inv);
  }
  __syncthreads();
  v8f o0 = {}, o1 = {};
#pragma unroll
  for (int k0 = 0; k0 < 64; k0 += 32) {           // O = P @ V (vts is B^T)
    v16h a  = frag_a(qs + sm * 64 + k0, 64);
    v16h b0 = frag_b(vts + sn * 64 + k0, 64);
    v16h b1 = frag_b(vts + (sn + 16) * 64 + k0, 64);
    o0 = wmma_f16(a, b0, o0);
    o1 = wmma_f16(a, b1, o1);
  }
  size_t ob = (size_t)win * 64 * CDIM + head * 64;
#pragma unroll
  for (int j = 0; j < 8; j++) {
    attout[ob + (size_t)(sm + mIn + j) * CDIM + sn + nIn]      = (_Float16)o0[j];
    attout[ob + (size_t)(sm + mIn + j) * CDIM + sn + 16 + nIn] = (_Float16)o1[j];
  }
}

// ---------------------------------------------------------------------------
extern "C" void kernel_launch(void* const* d_in, const int* in_sizes, int n_in,
                              void* d_out, int out_size, void* d_ws, size_t ws_size,
                              hipStream_t stream) {
  (void)in_sizes; (void)n_in; (void)out_size; (void)ws_size;
  const float* x      = (const float*)d_in[0];
  const float* ln1_g  = (const float*)d_in[1];
  const float* ln1_b  = (const float*)d_in[2];
  const float* qkv_w  = (const float*)d_in[3];
  const float* qkv_b  = (const float*)d_in[4];
  const float* proj_w = (const float*)d_in[5];
  const float* proj_b = (const float*)d_in[6];
  const float* ln2_g  = (const float*)d_in[7];
  const float* ln2_b  = (const float*)d_in[8];
  const float* w1     = (const float*)d_in[9];
  const float* b1     = (const float*)d_in[10];
  const float* w2     = (const float*)d_in[11];
  const float* b2     = (const float*)d_in[12];
  float* out = (float*)d_out;

  char* ws = (char*)d_ws;                          // ~454 MiB used
  _Float16* h16    = (_Float16*)(ws + 0);                  //  64 MiB (reused: h2)
  _Float16* qkv16  = (_Float16*)(ws + (64ull  << 20));     // 192 MiB
  _Float16* att16  = (_Float16*)(ws + (256ull << 20));     //  64 MiB
  float*    xo     = (float*)   (ws + (320ull << 20));     // 128 MiB
  _Float16* hid16  = (_Float16*)(ws + (64ull  << 20));     // 256 MiB (aliases qkv+att)
  _Float16* wqkvT  = (_Float16*)(ws + (448ull << 20));
  _Float16* wprojT = wqkvT  + 1536 * 512;
  _Float16* w1T    = wprojT + 512 * 512;
  _Float16* w2T    = w1T    + 512 * 2048;

  wconv_kernel<<<(512 * 1536 + 255) / 256, 256, 0, stream>>>(qkv_w, wqkvT, 512, 1536);
  wconv_kernel<<<(512 * 512  + 255) / 256, 256, 0, stream>>>(proj_w, wprojT, 512, 512);
  wconv_kernel<<<(512 * 2048 + 255) / 256, 256, 0, stream>>>(w1, w1T, 512, 2048);
  wconv_kernel<<<(2048 * 512 + 255) / 256, 256, 0, stream>>>(w2, w2T, 2048, 512);

  ln_kernel<<<TOKENS / 8, 256, 0, stream>>>(x, ln1_g, ln1_b, h16, 1);

  gemm_ep<0><<<dim3(1536 / 128, TOKENS / 128), 256, 0, stream>>>(
      h16, wqkvT, qkv_b, qkv16, 512, 1536, nullptr);

  attn_kernel<<<1024 * 8, 256, 0, stream>>>(qkv16, att16);

  gemm_ep<2><<<dim3(512 / 128, TOKENS / 128), 256, 0, stream>>>(
      att16, wprojT, proj_b, xo, 512, 512, x);

  ln_kernel<<<TOKENS / 8, 256, 0, stream>>>(xo, ln2_g, ln2_b, h16, 0);

  gemm_ep<1><<<dim3(HIDDIM / 128, TOKENS / 128), 256, 0, stream>>>(
      h16, w1T, b1, hid16, 512, HIDDIM, nullptr);

  gemm_ep<3><<<dim3(512 / 128, TOKENS / 128), 256, 0, stream>>>(
      hid16, w2T, b2, out, HIDDIM, 512, xo);
}
